// TransformerBlockMock_26491358281735
// MI455X (gfx1250) — matrix operand order repaired and checked by reading, not compiled, not run on hardware
//
#include <hip/hip_runtime.h>
#include <hip/hip_bf16.h>

typedef __bf16 bf16_t;
typedef __attribute__((ext_vector_type(8)))  __bf16 v8bf;
typedef __attribute__((ext_vector_type(16))) __bf16 v16bf;
typedef __attribute__((ext_vector_type(8)))  float  v8f;
typedef __attribute__((ext_vector_type(4)))  float  v4f;
typedef __attribute__((ext_vector_type(4)))  int    v4i;

constexpr int kN = 16384;   // total rows
constexpr int kH = 4096;    // hidden dim
constexpr int kM = 4;       // experts
constexpr int BM = 128, BN = 128, BK = 64;
constexpr int LDA = BK + 8;           // padded LDS row stride (elems): 144B rows, 16B aligned
constexpr int KSTEPS = kH / BK;       // 64

#if defined(__has_builtin)
#  if __has_builtin(__builtin_amdgcn_global_load_async_to_lds_b128)
#    define HAVE_ASYNC_LDS 1
#  endif
#endif

// Prototype revealed by clang diagnostics: param1 is a non-const int4* in the
// global (AS1, printed "__device__") address space; param2 is the LDS-side ptr.
typedef __attribute__((address_space(1))) v4i* gbl_v4ip;
typedef __attribute__((address_space(3))) v4i* lds_v4ip;

__device__ __forceinline__ void wait_async0() {
#if __has_builtin(__builtin_amdgcn_s_wait_asynccnt)
    __builtin_amdgcn_s_wait_asynccnt(0);
#else
    asm volatile("s_wait_asynccnt 0x0" ::: "memory");
#endif
}

union FragU { v16bf v; v8bf h[2]; };

__global__ __launch_bounds__(256)
void moe_norm_gemm_bf16(const float* __restrict__ x,
                        const float* __restrict__ nw,
                        const bf16_t* __restrict__ w,
                        float* __restrict__ out,
                        const int* __restrict__ g0, const int* __restrict__ g1,
                        const int* __restrict__ g2, const int* __restrict__ g3)
{
    __shared__ bf16_t sA[2][BM * LDA];
    __shared__ bf16_t sB[2][BN * LDA];
    __shared__ float  sS[kH];          // (norm_weights[e] + 1)

    const int tid   = threadIdx.x;
    const int lane  = tid & 31;
    const int wave  = tid >> 5;        // 0..7
    const int waveM = wave >> 1;       // 0..3  (32 rows each)
    const int waveN = wave & 1;        // 0..1  (64 cols each)
    const int lhalf = lane >> 4;       // 0/1
    const int l15   = lane & 15;

    const int row0 = blockIdx.y * BM;
    const int col0 = blockIdx.x * BN;

    // Expert for this block (group boundaries are multiples of BM in the
    // reference setup: g0..g3 == 4096).
    int gsz[4] = {*g0, *g1, *g2, *g3};
    int e = 0, off = 0;
    #pragma unroll
    for (int i = 0; i < kM; ++i) {
        if (row0 >= off && row0 < off + gsz[i]) e = i;
        off += gsz[i];
    }

    // Preload full per-column scale vector (nw[e] + 1) into LDS once.
    {
        const v4f* src = (const v4f*)(nw + (size_t)e * kH);
        v4f* dst = (v4f*)sS;
        for (int i = tid; i < kH / 4; i += 256) {
            v4f v = src[i];
            v[0] += 1.0f; v[1] += 1.0f; v[2] += 1.0f; v[3] += 1.0f;
            dst[i] = v;
        }
    }

    const bf16_t* wE = w + (size_t)e * kH * kH;

    v8f zero = {};
    v8f acc[2][4];
    #pragma unroll
    for (int mi = 0; mi < 2; ++mi)
        #pragma unroll
        for (int ni = 0; ni < 4; ++ni)
            acc[mi][ni] = zero;

    // ---- cooperative loader mapping: each thread owns a 32-element slice ----
    const int lr = tid >> 1;           // row within tile: 0..127
    const int lh = (tid & 1) * 32;     // K offset within tile: 0 or 32 elems

    const float*  gA = x  + (size_t)(row0 + lr) * kH + lh;   // + kt
    const bf16_t* gB = wE + (size_t)(col0 + lr) * kH + lh;   // + kt

    v4f aStage[8];

    auto loadA_g = [&](int kt) {
        const v4f* p = (const v4f*)(gA + kt);
        #pragma unroll
        for (int i = 0; i < 8; ++i) aStage[i] = p[i];
    };
    auto storeA_l = [&](int kt, int buf) {
        const v4f* s4 = (const v4f*)(sS + kt + lh);
        v8bf* dst = (v8bf*)(&sA[buf][lr * LDA + lh]);
        #pragma unroll
        for (int c = 0; c < 4; ++c) {
            v4f f0 = aStage[2 * c + 0], f1 = aStage[2 * c + 1];
            v4f s0 = s4[2 * c + 0],     s1 = s4[2 * c + 1];
            v8bf o;
            #pragma unroll
            for (int j = 0; j < 4; ++j) o[j]     = (bf16_t)(f0[j] * s0[j]);
            #pragma unroll
            for (int j = 0; j < 4; ++j) o[4 + j] = (bf16_t)(f1[j] * s1[j]);
            dst[c] = o;
        }
    };

#if defined(HAVE_ASYNC_LDS)
    // ASYNCcnt-tracked DMA: INST_OFFSET is added to BOTH the global and LDS
    // address, so immediate offsets 0/16/32/48 copy this thread's 64B slice.
    auto loadB_async = [&](int kt, int buf) {
        char* gm = const_cast<char*>((const char*)(gB + kt));
        char* lm = (char*)(&sB[buf][lr * LDA + lh]);
        gbl_v4ip gp = (gbl_v4ip)gm;
        lds_v4ip lp = (lds_v4ip)lm;
        __builtin_amdgcn_global_load_async_to_lds_b128(gp, lp, 0,  0);
        __builtin_amdgcn_global_load_async_to_lds_b128(gp, lp, 16, 0);
        __builtin_amdgcn_global_load_async_to_lds_b128(gp, lp, 32, 0);
        __builtin_amdgcn_global_load_async_to_lds_b128(gp, lp, 48, 0);
    };
#else
    v8bf bStage[4];
    auto loadB_g = [&](int kt) {
        const v8bf* p = (const v8bf*)(gB + kt);
        #pragma unroll
        for (int i = 0; i < 4; ++i) bStage[i] = p[i];
    };
    auto storeB_l = [&](int buf) {
        v8bf* dst = (v8bf*)(&sB[buf][lr * LDA + lh]);
        #pragma unroll
        for (int i = 0; i < 4; ++i) dst[i] = bStage[i];
    };
#endif

    auto compute = [&](int buf) {
        const bf16_t* Ab = sA[buf];
        const bf16_t* Bb = sB[buf];
        #pragma unroll
        for (int k0 = 0; k0 < BK; k0 += 32) {
            FragU aF[2];
            #pragma unroll
            for (int mi = 0; mi < 2; ++mi) {
                // A 16x32, contiguous k order chosen here (same as B): lane m holds K 0-15; lane m+16 holds K 16-31
                const bf16_t* pa = Ab + (waveM * 32 + mi * 16 + l15) * LDA + k0 + (lhalf << 4);
                aF[mi].h[0] = *(const v8bf*)(pa);
                aF[mi].h[1] = *(const v8bf*)(pa + 8);
            }
            #pragma unroll
            for (int ni = 0; ni < 4; ++ni) {
                // B 32x16, contiguous k order chosen here (same as A): lane n holds K 0-15; lane n+16 holds K 16-31
                const bf16_t* pb = Bb + (waveN * 64 + ni * 16 + l15) * LDA + k0 + (lhalf << 4);
                FragU bF;
                bF.h[0] = *(const v8bf*)(pb);
                bF.h[1] = *(const v8bf*)(pb + 8);
                #pragma unroll
                for (int mi = 0; mi < 2; ++mi) {
                    acc[mi][ni] = __builtin_amdgcn_wmma_f32_16x16x32_bf16(
                        false, aF[mi].v, false, bF.v,
                        (short)0, acc[mi][ni], false, false);
                }
            }
        }
    };

    // ---- software pipeline: double-buffered LDS, one barrier per K-step ----
#if defined(HAVE_ASYNC_LDS)
    loadB_async(0, 0);
#else
    loadB_g(0);
#endif
    loadA_g(0);
    __syncthreads();                   // sS visible before conversion below
    storeA_l(0, 0);
#if !defined(HAVE_ASYNC_LDS)
    storeB_l(0);
#endif
    wait_async0();
    __syncthreads();

    for (int s = 0; s < KSTEPS; ++s) {
        const int cur = s & 1, nxt = cur ^ 1;
        const int ktn = (s + 1) * BK;
        const bool more = (s + 1 < KSTEPS);
        if (more) {
#if defined(HAVE_ASYNC_LDS)
            loadB_async(ktn, nxt);     // DMA into the buffer all waves just vacated
#else
            loadB_g(ktn);
#endif
            loadA_g(ktn);              // global loads in flight across the WMMAs
        }
        compute(cur);
        if (more) {
            storeA_l(ktn, nxt);        // scale + bf16-round + LDS deposit
#if !defined(HAVE_ASYNC_LDS)
            storeB_l(nxt);
#endif
        }
        wait_async0();
        __syncthreads();
    }

    // ---- store C (f32): VGPR j: lanes 0-15 -> M=j, lanes 16-31 -> M=j+8; N = lane&15 ----
    #pragma unroll
    for (int mi = 0; mi < 2; ++mi) {
        #pragma unroll
        for (int ni = 0; ni < 4; ++ni) {
            const int r = row0 + waveM * 32 + mi * 16 + (lhalf << 3);
            const int c = col0 + waveN * 64 + ni * 16 + l15;
            float* po = out + (size_t)r * kH + c;
            #pragma unroll
            for (int j = 0; j < 8; ++j)
                po[(size_t)j * kH] = acc[mi][ni][j];
        }
    }
}

extern "C" void kernel_launch(void* const* d_in, const int* in_sizes, int n_in,
                              void* d_out, int out_size, void* d_ws, size_t ws_size,
                              hipStream_t stream) {
    const float*  x  = (const float*)d_in[0];
    const float*  nw = (const float*)d_in[1];
    const bf16_t* w  = (const bf16_t*)d_in[2];
    const int* g0 = (const int*)d_in[3];
    const int* g1 = (const int*)d_in[4];
    const int* g2 = (const int*)d_in[5];
    const int* g3 = (const int*)d_in[6];
    float* out = (float*)d_out;

    dim3 grid(kH / BN, kN / BM);   // (32, 128)
    dim3 block(256);
    moe_norm_gemm_bf16<<<grid, block, 0, stream>>>(x, nw, w, out, g0, g1, g2, g3);
}
